// EncodeSequential_74182675136896
// MI455X (gfx1250) — matrix operand order repaired and checked by reading, not compiled, not run on hardware
//
#include <hip/hip_runtime.h>
#include <hip/hip_bf16.h>

// Causal attention pipeline for MI455X (gfx1250, wave32, WMMA).
//   k1: QKV projection  (f32 -> f16 Q,K,V^T), 128x128 tile, reg-double-buffered
//   k2: S = Q K^T       (causal tile skip), async-to-LDS double buffered
//   k3: row softmax     (index-masked, zeros above diagonal)
//   k4: O = P V         (causal K-limit), async-to-LDS double buffered
// Workspace: 3*B*S*D f16 (48MB) + B*S*S f16 (32MB) = 80MB.

#define D_MODEL 1024
#define SEQ     2048
#define BATCH   4
#define LDT     40   // LDS tile row stride in halves (32 data + 8 pad, 80B rows)

typedef _Float16 v8h  __attribute__((ext_vector_type(8)));
typedef _Float16 v16h __attribute__((ext_vector_type(16)));
typedef float    v8f  __attribute__((ext_vector_type(8)));

union V16HU { v16h v; v8h h[2]; };
union H16U  { _Float16 h[16]; v8h v[2]; };

// ---- fragment loads from LDS (ISA 7.12.2 layouts, wave32) -------------------
// A 16x32 f16: lane%16 = M; lanes 0-15: v0-3 = K0..7, v4-7 = K16..23;
//              lanes 16-31: v0-3 = K8..15, v4-7 = K24..31.
__device__ __forceinline__ v16h load_a_frag(const _Float16* As, int mbase, int lane) {
    const int m    = mbase + (lane & 15);
    const int half = lane >> 4;
    V16HU a;
    a.h[0] = *reinterpret_cast<const v8h*>(As + m * LDT + half * 8);
    a.h[1] = *reinterpret_cast<const v8h*>(As + m * LDT + 16 + half * 8);
    return a.v;
}

// B 32x16 f16 (stored N-major in LDS): lane%16 = N;
// lanes 0-15: v0-3 = K0..7, v4-7 = K16..23; lanes 16-31: v0-3 = K8..15, v4-7 = K24..31.
__device__ __forceinline__ v16h load_b_frag(const _Float16* Bs, int nbase, int lane) {
    const int n    = nbase + (lane & 15);
    const int half = lane >> 4;
    V16HU b;
    b.h[0] = *reinterpret_cast<const v8h*>(Bs + n * LDT + half * 8);
    b.h[1] = *reinterpret_cast<const v8h*>(Bs + n * LDT + 16 + half * 8);
    return b.v;
}

// ---- per-wave 32x64 WMMA burst (2 A-frags x 4 B-frags = 8 WMMAs) ------------
__device__ __forceinline__ void mma_tile(const _Float16* As, const _Float16* Bs,
                                         int mbase, int nbase, int lane,
                                         v8f acc[2][4]) {
    v16h a0 = load_a_frag(As, mbase, lane);
    v16h a1 = load_a_frag(As, mbase + 16, lane);
#pragma unroll
    for (int t = 0; t < 4; ++t) {
        v16h b = load_b_frag(Bs, nbase + t * 16, lane);
        acc[0][t] = __builtin_amdgcn_wmma_f32_16x16x32_f16(false, a0, false, b,
                                                           (short)0, acc[0][t], false, false);
        acc[1][t] = __builtin_amdgcn_wmma_f32_16x16x32_f16(false, a1, false, b,
                                                           (short)0, acc[1][t], false, false);
    }
}

// ---- f32 staging through registers (for software pipelining) ----------------
struct FRegs { float4 f[4]; };

__device__ __forceinline__ FRegs gload_f32(const float* __restrict__ src, size_t ld, int tid) {
    const int r  = tid >> 1;
    const int c0 = (tid & 1) * 16;
    const float4* p = reinterpret_cast<const float4*>(src + (size_t)r * ld + c0);
    FRegs x;
    x.f[0] = p[0]; x.f[1] = p[1]; x.f[2] = p[2]; x.f[3] = p[3];
    return x;
}

__device__ __forceinline__ void store_cvt(_Float16* dst, const FRegs& x, int tid) {
    const int r  = tid >> 1;
    const int c0 = (tid & 1) * 16;
    H16U u;
#pragma unroll
    for (int j = 0; j < 4; ++j) {
        u.h[4 * j + 0] = (_Float16)x.f[j].x;
        u.h[4 * j + 1] = (_Float16)x.f[j].y;
        u.h[4 * j + 2] = (_Float16)x.f[j].z;
        u.h[4 * j + 3] = (_Float16)x.f[j].w;
    }
    *reinterpret_cast<v8h*>(dst + r * LDT + c0)     = u.v[0];
    *reinterpret_cast<v8h*>(dst + r * LDT + c0 + 8) = u.v[1];
}

// ---- async global->LDS staging (GLOBAL_LOAD_ASYNC_TO_LDS_B128, ASYNCcnt) ----
// 128x32 f16 tile, 256 threads: each thread issues two b128 async copies.
__device__ __forceinline__ void stage_async_f16(_Float16* dst, const _Float16* __restrict__ src,
                                                size_t ld, int tid) {
    const int r  = tid >> 1;
    const int c0 = (tid & 1) * 16;
    // low 32 bits of the flat shared-aperture address == group-relative LDS offset
    unsigned lds_off = (unsigned)(unsigned long long)(dst + r * LDT + c0);
    unsigned long long gaddr = (unsigned long long)(src + (size_t)r * ld + c0);
    asm volatile("global_load_async_to_lds_b128 %0, %1, off\n\t"
                 "global_load_async_to_lds_b128 %0, %1, off offset:16"
                 :: "v"(lds_off), "v"(gaddr) : "memory");
}

__device__ __forceinline__ void wait_async_le4() { asm volatile("s_wait_asynccnt 0x4" ::: "memory"); }
__device__ __forceinline__ void wait_async_0()   { asm volatile("s_wait_asynccnt 0x0" ::: "memory"); }

// ---- kernel 1: out[m,n] = (sum_d X[m,d]*W[n,d] + bias[n]) * scale -----------
// transposeOut=1 stores per-batch transposed: vT[b][n][s].
__global__ __launch_bounds__(256) void proj128(const float* __restrict__ X,
                                               const float* __restrict__ W,
                                               const float* __restrict__ bias,
                                               _Float16* __restrict__ out,
                                               float scale, int transposeOut) {
    __shared__ _Float16 As[128 * LDT];
    __shared__ _Float16 Bs[128 * LDT];
    const int tid  = threadIdx.x;
    const int lane = tid & 31;
    const int wave = tid >> 5;
    const int n0   = blockIdx.x * 128;
    const int m0   = blockIdx.y * 128;
    const int mb   = (wave >> 1) * 32;
    const int nb   = (wave & 1) * 64;

    v8f acc[2][4] = {};
    FRegs ra = gload_f32(X + (size_t)m0 * D_MODEL, D_MODEL, tid);
    FRegs rb = gload_f32(W + (size_t)n0 * D_MODEL, D_MODEL, tid);

    for (int kk = 0; kk < D_MODEL; kk += 32) {
        store_cvt(As, ra, tid);
        store_cvt(Bs, rb, tid);
        __syncthreads();
        if (kk + 32 < D_MODEL) {  // issue next K-slab loads before the WMMA burst
            ra = gload_f32(X + (size_t)m0 * D_MODEL + kk + 32, D_MODEL, tid);
            rb = gload_f32(W + (size_t)n0 * D_MODEL + kk + 32, D_MODEL, tid);
            if (kk + 64 < D_MODEL) {
                __builtin_prefetch(X + (size_t)(m0 + (tid >> 1)) * D_MODEL + kk + 64, 0, 3);
                __builtin_prefetch(W + (size_t)(n0 + (tid >> 1)) * D_MODEL + kk + 64, 0, 3);
            }
        }
        mma_tile(As, Bs, mb, nb, lane, acc);
        __syncthreads();
    }

    const int half = lane >> 4;
    const int nl   = lane & 15;
#pragma unroll
    for (int am = 0; am < 2; ++am) {
#pragma unroll
        for (int t = 0; t < 4; ++t) {
            const int n = n0 + nb + t * 16 + nl;
            const float bv = bias[n];
#pragma unroll
            for (int i = 0; i < 8; ++i) {
                const int m = m0 + mb + am * 16 + half * 8 + i;
                const float v = (acc[am][t][i] + bv) * scale;
                if (!transposeOut) {
                    out[(size_t)m * D_MODEL + n] = (_Float16)v;
                } else {
                    const int b = m / SEQ, s = m % SEQ;
                    out[((size_t)b * D_MODEL + n) * SEQ + s] = (_Float16)v;
                }
            }
        }
    }
}

// ---- generic async double-buffered f16 GEMM body ----------------------------
// Computes acc += A[128 x kiters*32] * B^T tile, A rows at asrc, B rows at bsrc.
__device__ __forceinline__ void gemm_f16_async(const _Float16* __restrict__ asrc, size_t alda,
                                               const _Float16* __restrict__ bsrc, size_t bldb,
                                               int kiters, _Float16* As0, _Float16* As1,
                                               _Float16* Bs0, _Float16* Bs1,
                                               int mb, int nb, int tid, int lane,
                                               v8f acc[2][4]) {
    _Float16* Abuf[2] = {As0, As1};
    _Float16* Bbuf[2] = {Bs0, Bs1};
    stage_async_f16(Abuf[0], asrc, alda, tid);
    stage_async_f16(Bbuf[0], bsrc, bldb, tid);
    int p = 0;
    for (int it = 0; it < kiters; ++it) {
        if (it + 1 < kiters) {
            stage_async_f16(Abuf[p ^ 1], asrc + (size_t)(it + 1) * 32, alda, tid);
            stage_async_f16(Bbuf[p ^ 1], bsrc + (size_t)(it + 1) * 32, bldb, tid);
            wait_async_le4();  // in-order completion: current stage has landed
        } else {
            wait_async_0();
        }
        __syncthreads();
        mma_tile(Abuf[p], Bbuf[p], mb, nb, lane, acc);
        __syncthreads();
        p ^= 1;
    }
}

// ---- kernel 2: S[b,q,j] = sum_d Q[b,q,d]*K[b,j,d]  (Q pre-scaled) -----------
__global__ __launch_bounds__(256) void scores128(const _Float16* __restrict__ Q,
                                                 const _Float16* __restrict__ K,
                                                 _Float16* __restrict__ S) {
    if (blockIdx.x > blockIdx.y) return;  // tile fully above causal diagonal
    __shared__ _Float16 As[2][128 * LDT];
    __shared__ _Float16 Bs[2][128 * LDT];
    const int tid  = threadIdx.x;
    const int lane = tid & 31;
    const int wave = tid >> 5;
    const int bz   = blockIdx.z;
    const int j0   = blockIdx.x * 128;
    const int q0   = blockIdx.y * 128;
    const int mb   = (wave >> 1) * 32;
    const int nb   = (wave & 1) * 64;
    const _Float16* Qb = Q + (size_t)bz * SEQ * D_MODEL + (size_t)q0 * D_MODEL;
    const _Float16* Kb = K + (size_t)bz * SEQ * D_MODEL + (size_t)j0 * D_MODEL;

    v8f acc[2][4] = {};
    gemm_f16_async(Qb, D_MODEL, Kb, D_MODEL, D_MODEL / 32,
                   As[0], As[1], Bs[0], Bs[1], mb, nb, tid, lane, acc);

    _Float16* Sb = S + (size_t)bz * SEQ * SEQ;
    const int half = lane >> 4;
    const int nl   = lane & 15;
#pragma unroll
    for (int am = 0; am < 2; ++am) {
#pragma unroll
        for (int t = 0; t < 4; ++t) {
            const int j = j0 + nb + t * 16 + nl;
#pragma unroll
            for (int i = 0; i < 8; ++i) {
                const int q = q0 + mb + am * 16 + half * 8 + i;
                Sb[(size_t)q * SEQ + j] = (_Float16)acc[am][t][i];
            }
        }
    }
}

// ---- kernel 3: causal softmax over one row; writes P (zeros for j>q) --------
__global__ __launch_bounds__(256) void softmax_row(_Float16* __restrict__ S) {
    __shared__ float red[256];
    const int row = blockIdx.x;
    const int b = row / SEQ, q = row % SEQ;
    _Float16* p = S + (size_t)b * SEQ * SEQ + (size_t)q * SEQ;
    const int n = q + 1;
    const int tid = threadIdx.x;

    float mx = -1e30f;
    for (int j = tid; j < n; j += 256) mx = fmaxf(mx, (float)p[j]);
    red[tid] = mx;
    __syncthreads();
    for (int s = 128; s > 0; s >>= 1) {
        if (tid < s) red[tid] = fmaxf(red[tid], red[tid + s]);
        __syncthreads();
    }
    mx = red[0];
    __syncthreads();

    float sum = 0.0f;
    for (int j = tid; j < n; j += 256) sum += __expf((float)p[j] - mx);
    red[tid] = sum;
    __syncthreads();
    for (int s = 128; s > 0; s >>= 1) {
        if (tid < s) red[tid] += red[tid + s];
        __syncthreads();
    }
    const float inv = 1.0f / red[0];

    for (int j = tid; j < SEQ; j += 256) {
        const float v = (j < n) ? __expf((float)p[j] - mx) * inv : 0.0f;
        p[j] = (_Float16)v;
    }
}

// ---- kernel 4: O[b,q,d] = sum_j P[b,q,j]*V[b,j,d] using V^T rows ------------
__global__ __launch_bounds__(256) void pv128(const _Float16* __restrict__ P,
                                             const _Float16* __restrict__ VT,
                                             float* __restrict__ O) {
    __shared__ _Float16 As[2][128 * LDT];
    __shared__ _Float16 Bs[2][128 * LDT];
    const int tid  = threadIdx.x;
    const int lane = tid & 31;
    const int wave = tid >> 5;
    const int bz   = blockIdx.z;
    const int n0   = blockIdx.x * 128;  // output feature dim
    const int q0   = blockIdx.y * 128;  // query tile
    const int mb   = (wave >> 1) * 32;
    const int nb   = (wave & 1) * 64;
    const _Float16* Pb = P + (size_t)bz * SEQ * SEQ + (size_t)q0 * SEQ;
    const _Float16* Vb = VT + (size_t)bz * D_MODEL * SEQ + (size_t)n0 * SEQ;
    const int kiters = (q0 + 128) / 32;  // causal: P is zero for j > q

    v8f acc[2][4] = {};
    gemm_f16_async(Pb, SEQ, Vb, SEQ, kiters,
                   As[0], As[1], Bs[0], Bs[1], mb, nb, tid, lane, acc);

    const int half = lane >> 4;
    const int nl   = lane & 15;
#pragma unroll
    for (int am = 0; am < 2; ++am) {
#pragma unroll
        for (int t = 0; t < 4; ++t) {
            const int d = n0 + nb + t * 16 + nl;
#pragma unroll
            for (int i = 0; i < 8; ++i) {
                const int q = q0 + mb + am * 16 + half * 8 + i;
                O[((size_t)bz * SEQ + q) * D_MODEL + d] = acc[am][t][i];
            }
        }
    }
}

extern "C" void kernel_launch(void* const* d_in, const int* in_sizes, int n_in,
                              void* d_out, int out_size, void* d_ws, size_t ws_size,
                              hipStream_t stream) {
    const float* x  = (const float*)d_in[0];
    const float* Wq = (const float*)d_in[1];
    const float* bq = (const float*)d_in[2];
    const float* Wk = (const float*)d_in[3];
    const float* bk = (const float*)d_in[4];
    const float* Wv = (const float*)d_in[5];
    const float* bv = (const float*)d_in[6];
    float* out = (float*)d_out;

    _Float16* ws = (_Float16*)d_ws;
    const size_t qkv = (size_t)BATCH * SEQ * D_MODEL;
    _Float16* q_h = ws;
    _Float16* k_h = ws + qkv;
    _Float16* vT  = ws + 2 * qkv;
    _Float16* Sb  = ws + 3 * qkv;  // B*S*S f16

    const dim3 blk(256);
    const dim3 g1(D_MODEL / 128, (BATCH * SEQ) / 128);
    proj128<<<g1, blk, 0, stream>>>(x, Wq, bq, q_h, 0.03125f /*1/sqrt(1024)*/, 0);
    proj128<<<g1, blk, 0, stream>>>(x, Wk, bk, k_h, 1.0f, 0);
    proj128<<<g1, blk, 0, stream>>>(x, Wv, bv, vT, 1.0f, 1);

    const dim3 g2(SEQ / 128, SEQ / 128, BATCH);
    scores128<<<g2, blk, 0, stream>>>(q_h, k_h, Sb);

    softmax_row<<<dim3(BATCH * SEQ), dim3(256), 0, stream>>>(Sb);

    const dim3 g4(D_MODEL / 128, SEQ / 128, BATCH);
    pv128<<<g4, blk, 0, stream>>>(Sb, vT, out);
}
